// EGNNNetwork_56934086476467
// MI455X (gfx1250) — matrix operand order repaired and checked by reading, not compiled, not run on hardware
//
#include <hip/hip_runtime.h>
#include <hip/hip_bf16.h>

typedef __attribute__((ext_vector_type(16))) _Float16 v16h;
typedef __attribute__((ext_vector_type(8)))  _Float16 v8h;
typedef __attribute__((ext_vector_type(2)))  _Float16 v2h;
typedef __attribute__((ext_vector_type(8)))  float    v8f;

#define NN     50000
#define EE     800000
#define FEATS  64
#define EATTR  16
#define GG     64
#define OUTD   32
#define EIN    145
#define KP1    160      // K pad for edge GEMM1 (145 -> 160, 5 k-tiles)
#define H1     290      // 2*EIN
#define H1P    304      // N pad for edge GEMM1 (19 n-tiles)
#define H1K    320      // K pad for edge GEMM2 (10 k-tiles)
#define MD     64
#define CHID   256
#define NK     128
#define NH     128
#define GHID   128

#define DEV static __device__ __forceinline__

// ---- WMMA fragment helpers (layouts per CDNA5 ISA 7.12.2, wave32) ----

// A: 16x32 f16 tile from LDS row-major [16][ld]; two ds_load_b128 per lane
DEV v16h ld_a_lds(const _Float16* h, int ld, int kt) {
  const int lane = threadIdx.x & 31;
  const int row  = lane & 15;
  const int koff = kt * 32 + ((lane >> 4) << 3);   // 0 or 8
  const v8h* p = (const v8h*)(h + row * ld + koff);
  const v8h lo = p[0];        // K koff..koff+7
  const v8h hi = p[2];        // K koff+16..koff+23
  return __builtin_shufflevector(lo, hi, 0, 1, 2, 3, 4, 5, 6, 7,
                                 8, 9, 10, 11, 12, 13, 14, 15);
}

// B: 32x16 f16 tile from fragment-packed global weights.
// Packed layout: fragment (kt,nt) occupies 32 lanes x 16 halves contiguously.
DEV v16h ld_b_pack(const _Float16* wp, int ntiles, int kt, int nt) {
  const int lane = threadIdx.x & 31;
  return *(const v16h*)(wp + ((((kt * ntiles + nt) << 5) + lane) << 4));
}

DEV float silu(float v) { return v / (1.f + __expf(-v)); }

// D (16x16 f32) -> LDS f16 with bias + activation (act: 1=SiLU, 0=ReLU)
DEV void st_d_act_lds(_Float16* out, int ld, int nt, v8f acc,
                      const float* bias, int nvalid, int act) {
  const int lane = threadIdx.x & 31;
  const int col  = nt * 16 + (lane & 15);
  const float bb = (col < nvalid) ? bias[col] : 0.f;
#pragma unroll
  for (int i = 0; i < 8; ++i) {
    const int row = i + ((lane >> 4) << 3);
    float v = acc[i] + bb;
    v = act ? silu(v) : fmaxf(v, 0.f);
    out[row * ld + col] = (_Float16)v;
  }
}

// ---- weight conversion: f32 [K][N] row-major -> f16 fragment-packed ----
// dst index i decodes as: j = half-within-lane, l = lane, then (kt, nt).
__global__ void cvt_w_pack(const float* __restrict__ src, _Float16* __restrict__ dst,
                           int K, int Nc, int Kp, int Np) {
  const int i = blockIdx.x * blockDim.x + threadIdx.x;
  if (i >= Kp * Np) return;
  const int j      = i & 15;
  const int l      = (i >> 4) & 31;
  const int rest   = i >> 9;
  const int ntiles = Np >> 4;
  const int nt     = rest % ntiles;
  const int kt     = rest / ntiles;
  const int col    = nt * 16 + (l & 15);
  const int k      = kt * 32 + ((l >> 4) << 3) + (j & 7) + ((j >> 3) << 4);
  const float v = (k < K && col < Nc) ? src[k * Nc + col] : 0.f;
  dst[i] = (_Float16)v;
}

__global__ void init_nodes(const float* __restrict__ x, const float* __restrict__ pos,
                           float* fa, _Float16* fh, float* ca) {
  int i = blockIdx.x * blockDim.x + threadIdx.x;
  if (i < NN * FEATS) { float v = x[i]; fa[i] = v; fh[i] = (_Float16)v; }
  if (i < NN * 3) ca[i] = pos[i];
}

// ---- fused edge kernel: one wave per 16-edge tile ----
__global__ __launch_bounds__(32)
void edge_kernel(const int* __restrict__ ei, const float* __restrict__ ea,
                 const _Float16* __restrict__ fh, const float* __restrict__ coors,
                 const _Float16* __restrict__ ew1, const float* __restrict__ eb1,
                 const _Float16* __restrict__ ew2, const float* __restrict__ eb2,
                 const _Float16* __restrict__ cw1, const float* __restrict__ cb1,
                 const float* __restrict__ cw2, const float* __restrict__ cb2,
                 float* __restrict__ m_acc, float* __restrict__ coor_acc) {
  __shared__ __align__(16) _Float16 hbuf[16][KP1];
  __shared__ __align__(16) _Float16 hid[16][H1K];
  __shared__ __align__(16) _Float16 mt[16][MD];
  __shared__ __align__(16) _Float16 chb[16][CHID];
  __shared__ float    rel[16][3];
  __shared__ float    cwpart[32];
  __shared__ int      sdst[16], ssrc[16];

  const int lane = threadIdx.x;
  const int e0   = blockIdx.x * 16;

  if (lane == 0) { __builtin_prefetch(ew1, 0, 3); __builtin_prefetch(ew2, 0, 3); }

  if (lane < 16) {
    const int e = e0 + lane;
    const int s = ei[e];
    const int d = ei[EE + e];
    ssrc[lane] = s; sdst[lane] = d;
    const float rx = coors[(long)s * 3 + 0] - coors[(long)d * 3 + 0];
    const float ry = coors[(long)s * 3 + 1] - coors[(long)d * 3 + 1];
    const float rz = coors[(long)s * 3 + 2] - coors[(long)d * 3 + 2];
    rel[lane][0] = rx; rel[lane][1] = ry; rel[lane][2] = rz;
    hbuf[lane][128 + EATTR] = (_Float16)(rx * rx + ry * ry + rz * rz); // col 144
  }
  __syncthreads();

  // gather: h = [feats[dst](64) | feats[src](64) | edge_attr(16) | dist(1) | pad(15)]
  {
    const int c2 = lane * 2;   // 2 contiguous halves per lane -> 4B packed ops
    for (int r = 0; r < 16; ++r) {
      const long s = ssrc[r], d = sdst[r];
      *(v2h*)&hbuf[r][c2]      = *(const v2h*)(fh + d * 64 + c2);
      *(v2h*)&hbuf[r][64 + c2] = *(const v2h*)(fh + s * 64 + c2);
      if (lane < EATTR) hbuf[r][128 + lane] = (_Float16)ea[(long)(e0 + r) * EATTR + lane];
      if (lane >= 17)   hbuf[r][128 + lane] = (_Float16)0.f;   // cols 145..159
    }
  }
  // zero the K-pad region of hid (cols 304..319)
  for (int idx = lane; idx < 16 * (H1K - H1P); idx += 32) {
    const int r = idx >> 4, c = H1P + (idx & 15);
    hid[r][c] = (_Float16)0.f;
  }
  __syncthreads();

  // GEMM1: (16x160) @ (160x304) + b, SiLU -> hid
  {
    v16h aF[KP1 / 32];
#pragma unroll
    for (int kt = 0; kt < KP1 / 32; ++kt) aF[kt] = ld_a_lds(&hbuf[0][0], KP1, kt);
    for (int nt = 0; nt < H1P / 16; ++nt) {
      v8f acc = {};
#pragma unroll
      for (int kt = 0; kt < KP1 / 32; ++kt) {
        v16h b = ld_b_pack(ew1, H1P / 16, kt, nt);
        acc = __builtin_amdgcn_wmma_f32_16x16x32_f16(false, aF[kt], false, b, (short)0, acc, false, false);
      }
      st_d_act_lds(&hid[0][0], H1K, nt, acc, eb1, H1, 1);
    }
  }
  __syncthreads();

  // GEMM2: (16x320) @ (320x64) + b, SiLU -> m tile; scatter m to m_acc[dst]
  {
    v16h aF[H1K / 32];
#pragma unroll
    for (int kt = 0; kt < H1K / 32; ++kt) aF[kt] = ld_a_lds(&hid[0][0], H1K, kt);
#pragma unroll
    for (int nt = 0; nt < MD / 16; ++nt) {
      v8f acc = {};
#pragma unroll
      for (int kt = 0; kt < H1K / 32; ++kt) {
        v16h b = ld_b_pack(ew2, MD / 16, kt, nt);
        acc = __builtin_amdgcn_wmma_f32_16x16x32_f16(false, aF[kt], false, b, (short)0, acc, false, false);
      }
      const int col = nt * 16 + (lane & 15);
      const float bb = eb2[col];
#pragma unroll
      for (int i = 0; i < 8; ++i) {
        const int row = i + ((lane >> 4) << 3);
        float v = silu(acc[i] + bb);
        mt[row][col] = (_Float16)v;
        atomicAdd(&m_acc[(long)sdst[row] * MD + col], v);
      }
    }
  }
  __syncthreads();

  // GEMM3: (16x64) @ (64x256) + b, SiLU -> chb
  {
    v16h aF[MD / 32];
#pragma unroll
    for (int kt = 0; kt < MD / 32; ++kt) aF[kt] = ld_a_lds(&mt[0][0], MD, kt);
    for (int nt = 0; nt < CHID / 16; ++nt) {
      v8f acc = {};
#pragma unroll
      for (int kt = 0; kt < MD / 32; ++kt) {
        v16h b = ld_b_pack(cw1, CHID / 16, kt, nt);
        acc = __builtin_amdgcn_wmma_f32_16x16x32_f16(false, aF[kt], false, b, (short)0, acc, false, false);
      }
      st_d_act_lds(&chb[0][0], CHID, nt, acc, cb1, CHID, 1);
    }
  }
  __syncthreads();

  // cw = chb . c_w2 + c_b2 (split-K across half-waves); coord scatter
  {
    const int e = lane & 15, hh = lane >> 4;
    float s = 0.f;
    const float* w2 = cw2 + hh * 128;
    for (int k = 0; k < 128; ++k) s += (float)chb[e][hh * 128 + k] * w2[k];
    cwpart[lane] = s;
  }
  __syncthreads();
  if (lane < 16) {
    const float s = cwpart[lane] + cwpart[lane + 16] + cb2[0];
    const long d = sdst[lane];
    atomicAdd(&coor_acc[d * 3 + 0], s * rel[lane][0]);
    atomicAdd(&coor_acc[d * 3 + 1], s * rel[lane][1]);
    atomicAdd(&coor_acc[d * 3 + 2], s * rel[lane][2]);
  }
}

// ---- global mean/var reduction over feats ----
__global__ void reduce_stats(const float* __restrict__ f, float* __restrict__ stats) {
  __shared__ float s1[256], s2[256];
  const int tid = threadIdx.x;
  float a = 0.f, b = 0.f;
  for (long i = (long)blockIdx.x * 256 + tid; i < (long)NN * FEATS; i += (long)gridDim.x * 256) {
    float v = f[i]; a += v; b += v * v;
  }
  s1[tid] = a; s2[tid] = b; __syncthreads();
  for (int st = 128; st > 0; st >>= 1) {
    if (tid < st) { s1[tid] += s1[tid + st]; s2[tid] += s2[tid + st]; }
    __syncthreads();
  }
  if (tid == 0) { atomicAdd(&stats[0], s1[0]); atomicAdd(&stats[1], s2[0]); }
}

// ---- node update: one wave per 16-node tile ----
__global__ __launch_bounds__(32)
void node_kernel(const float* __restrict__ fin, const float* __restrict__ m_acc,
                 const float* __restrict__ coor_acc, const float* __restrict__ cin,
                 const float* __restrict__ stats,
                 const float* __restrict__ lnw, const float* __restrict__ lnb,
                 const _Float16* __restrict__ nw1, const float* __restrict__ nb1,
                 const _Float16* __restrict__ nw2, const float* __restrict__ nb2,
                 float* __restrict__ fout, _Float16* __restrict__ fhout,
                 float* __restrict__ cout) {
  __shared__ __align__(16) _Float16 xin[16][NK];
  __shared__ __align__(16) _Float16 hid[16][NH];
  const int lane = threadIdx.x;
  const int n0   = blockIdx.x * 16;
  const float inv  = 1.f / ((float)NN * FEATS);
  const float mu   = stats[0] * inv;
  const float var  = stats[1] * inv - mu * mu;
  const float istd = 1.f / (sqrtf(fmaxf(var, 0.f)) + 1e-5f);

  for (int r = 0; r < 16; ++r) {
    const long node = n0 + r;
    for (int c = lane; c < 64; c += 32) {
      float v = (fin[node * 64 + c] - mu) * istd * lnw[c] + lnb[c];
      xin[r][c]      = (_Float16)v;
      xin[r][64 + c] = (_Float16)m_acc[node * 64 + c];
    }
  }
  __syncthreads();

  {
    v16h aF[NK / 32];
#pragma unroll
    for (int kt = 0; kt < NK / 32; ++kt) aF[kt] = ld_a_lds(&xin[0][0], NK, kt);
#pragma unroll
    for (int nt = 0; nt < NH / 16; ++nt) {
      v8f acc = {};
#pragma unroll
      for (int kt = 0; kt < NK / 32; ++kt) {
        v16h b = ld_b_pack(nw1, NH / 16, kt, nt);
        acc = __builtin_amdgcn_wmma_f32_16x16x32_f16(false, aF[kt], false, b, (short)0, acc, false, false);
      }
      st_d_act_lds(&hid[0][0], NH, nt, acc, nb1, NH, 1);
    }
  }
  __syncthreads();

  {
    v16h aF[NH / 32];
#pragma unroll
    for (int kt = 0; kt < NH / 32; ++kt) aF[kt] = ld_a_lds(&hid[0][0], NH, kt);
#pragma unroll
    for (int nt = 0; nt < FEATS / 16; ++nt) {
      v8f acc = {};
#pragma unroll
      for (int kt = 0; kt < NH / 32; ++kt) {
        v16h b = ld_b_pack(nw2, FEATS / 16, kt, nt);
        acc = __builtin_amdgcn_wmma_f32_16x16x32_f16(false, aF[kt], false, b, (short)0, acc, false, false);
      }
      const int col = nt * 16 + (lane & 15);
      const float bb = nb2[col];
#pragma unroll
      for (int i = 0; i < 8; ++i) {
        const int row = i + ((lane >> 4) << 3);
        const long node = n0 + row;
        float v = fin[node * 64 + col] + acc[i] + bb;  // residual
        fout[node * 64 + col]  = v;
        fhout[node * 64 + col] = (_Float16)v;
      }
    }
  }
  for (int idx = lane; idx < 48; idx += 32) {
    const long a = (long)n0 * 3 + idx;
    cout[a] = cin[a] + coor_acc[a];
  }
}

// ---- pooling ----
__global__ void pool_accum(const float* __restrict__ f, const int* __restrict__ batch,
                           float* __restrict__ psum, float* __restrict__ pcnt) {
  const int i = blockIdx.x * blockDim.x + threadIdx.x;
  if (i >= NN) return;
  const int g = batch[i];
  atomicAdd(&pcnt[g], 1.f);
  for (int c = 0; c < FEATS; ++c) atomicAdd(&psum[g * FEATS + c], f[(long)i * FEATS + c]);
}

// ---- final MLP: one block (4 waves), WMMA ----
__global__ __launch_bounds__(128)
void gmlp_kernel(const float* __restrict__ psum, const float* __restrict__ pcnt,
                 const _Float16* __restrict__ gw1, const float* __restrict__ gb1,
                 const _Float16* __restrict__ gw2, const float* __restrict__ gb2,
                 float* __restrict__ out) {
  __shared__ __align__(16) _Float16 pool[GG][FEATS];
  __shared__ __align__(16) _Float16 hid[GG][GHID];
  const int tid = threadIdx.x;
  for (int i = tid; i < GG * FEATS; i += 128) {
    const int g = i >> 6;
    const float c = fmaxf(pcnt[g], 1.f);
    pool[g][i & 63] = (_Float16)(psum[i] / c);
  }
  __syncthreads();
  const int wave = tid >> 5;
  {
    v16h aF[FEATS / 32];
#pragma unroll
    for (int kt = 0; kt < FEATS / 32; ++kt) aF[kt] = ld_a_lds(&pool[wave * 16][0], FEATS, kt);
    for (int nt = 0; nt < GHID / 16; ++nt) {
      v8f acc = {};
#pragma unroll
      for (int kt = 0; kt < FEATS / 32; ++kt) {
        v16h b = ld_b_pack(gw1, GHID / 16, kt, nt);
        acc = __builtin_amdgcn_wmma_f32_16x16x32_f16(false, aF[kt], false, b, (short)0, acc, false, false);
      }
      st_d_act_lds(&hid[wave * 16][0], GHID, nt, acc, gb1, GHID, 0 /*relu*/);
    }
  }
  __syncthreads();
  {
    v16h aF[GHID / 32];
#pragma unroll
    for (int kt = 0; kt < GHID / 32; ++kt) aF[kt] = ld_a_lds(&hid[wave * 16][0], GHID, kt);
#pragma unroll
    for (int nt = 0; nt < OUTD / 16; ++nt) {
      v8f acc = {};
#pragma unroll
      for (int kt = 0; kt < GHID / 32; ++kt) {
        v16h b = ld_b_pack(gw2, OUTD / 16, kt, nt);
        acc = __builtin_amdgcn_wmma_f32_16x16x32_f16(false, aF[kt], false, b, (short)0, acc, false, false);
      }
      const int lane = tid & 31;
      const int col  = nt * 16 + (lane & 15);
      const float bb = gb2[col];
#pragma unroll
      for (int i = 0; i < 8; ++i) {
        const int row = wave * 16 + i + ((lane >> 4) << 3);
        out[row * OUTD + col] = acc[i] + bb;
      }
    }
  }
}

// ---------------- host orchestration ----------------
static inline void* take(char*& cur, size_t bytes) {
  void* p = (void*)cur;
  cur += (bytes + 255) & ~(size_t)255;
  return p;
}

extern "C" void kernel_launch(void* const* d_in, const int* in_sizes, int n_in,
                              void* d_out, int out_size, void* d_ws, size_t ws_size,
                              hipStream_t stream) {
  (void)in_sizes; (void)n_in; (void)out_size; (void)ws_size;
  const float* x   = (const float*)d_in[0];
  const float* pos = (const float*)d_in[1];
  const int*   ei  = (const int*)d_in[2];
  const float* ea  = (const float*)d_in[3];
  const int*   bat = (const int*)d_in[4];
  const float* P[2][14];
  for (int i = 0; i < 14; ++i) { P[0][i] = (const float*)d_in[5 + i]; P[1][i] = (const float*)d_in[19 + i]; }
  const float* GM[4];
  for (int i = 0; i < 4; ++i) GM[i] = (const float*)d_in[33 + i];
  float* out = (float*)d_out;

  // workspace layout
  char* cur = (char*)d_ws;
  float*    feats_a  = (float*)take(cur, (size_t)NN * FEATS * 4);
  float*    feats_b  = (float*)take(cur, (size_t)NN * FEATS * 4);
  _Float16* feats_h  = (_Float16*)take(cur, (size_t)NN * FEATS * 2);
  float*    coors_a  = (float*)take(cur, (size_t)NN * 3 * 4);
  float*    coors_b  = (float*)take(cur, (size_t)NN * 3 * 4);
  float*    m_acc    = (float*)take(cur, (size_t)NN * FEATS * 4);
  float*    coor_acc = (float*)take(cur, (size_t)NN * 3 * 4);
  float*    stats    = (float*)take(cur, 256);
  float*    psum     = (float*)take(cur, (size_t)GG * FEATS * 4);
  float*    pcnt     = (float*)take(cur, (size_t)GG * 4);
  _Float16 *ew1h[2], *ew2h[2], *cw1h[2], *nw1h[2], *nw2h[2];
  for (int l = 0; l < 2; ++l) {
    ew1h[l] = (_Float16*)take(cur, (size_t)KP1 * H1P * 2);
    ew2h[l] = (_Float16*)take(cur, (size_t)H1K * MD * 2);
    cw1h[l] = (_Float16*)take(cur, (size_t)MD * CHID * 2);
    nw1h[l] = (_Float16*)take(cur, (size_t)NK * NH * 2);
    nw2h[l] = (_Float16*)take(cur, (size_t)NH * FEATS * 2);
  }
  _Float16* gw1h = (_Float16*)take(cur, (size_t)FEATS * GHID * 2);
  _Float16* gw2h = (_Float16*)take(cur, (size_t)GHID * OUTD * 2);

  auto cvt = [&](const float* s, _Float16* d, int K, int Nc, int Kp, int Np) {
    int tot = Kp * Np;
    cvt_w_pack<<<(tot + 255) / 256, 256, 0, stream>>>(s, d, K, Nc, Kp, Np);
  };
  for (int l = 0; l < 2; ++l) {
    cvt(P[l][0],  ew1h[l], EIN, H1,   KP1, H1P);   // e_w1
    cvt(P[l][2],  ew2h[l], H1,  MD,   H1K, MD);    // e_w2
    cvt(P[l][10], cw1h[l], MD,  CHID, MD,  CHID);  // c_w1
    cvt(P[l][6],  nw1h[l], NK,  NH,   NK,  NH);    // n_w1
    cvt(P[l][8],  nw2h[l], NH,  FEATS,NH,  FEATS); // n_w2
  }
  cvt(GM[0], gw1h, FEATS, GHID, FEATS, GHID);
  cvt(GM[2], gw2h, GHID, OUTD, GHID, OUTD);

  init_nodes<<<(NN * FEATS + 255) / 256, 256, 0, stream>>>(x, pos, feats_a, feats_h, coors_a);

  float* fin  = feats_a;  float* fout = feats_b;
  float* cin  = coors_a;  float* cout = coors_b;
  for (int l = 0; l < 2; ++l) {
    hipMemsetAsync(m_acc, 0, (size_t)NN * FEATS * 4, stream);
    hipMemsetAsync(coor_acc, 0, (size_t)NN * 3 * 4, stream);
    hipMemsetAsync(stats, 0, 8, stream);
    edge_kernel<<<EE / 16, 32, 0, stream>>>(
        ei, ea, feats_h, cin,
        ew1h[l], P[l][1], ew2h[l], P[l][3],
        cw1h[l], P[l][11], P[l][12], P[l][13],
        m_acc, coor_acc);
    reduce_stats<<<1024, 256, 0, stream>>>(fin, stats);
    node_kernel<<<NN / 16, 32, 0, stream>>>(
        fin, m_acc, coor_acc, cin, stats,
        P[l][4], P[l][5], nw1h[l], P[l][7], nw2h[l], P[l][9],
        fout, feats_h, cout);
    float* t;
    t = fin; fin = fout; fout = t;
    t = cin; cin = cout; cout = t;
  }

  hipMemsetAsync(psum, 0, (size_t)GG * FEATS * 4, stream);
  hipMemsetAsync(pcnt, 0, (size_t)GG * 4, stream);
  pool_accum<<<(NN + 255) / 256, 256, 0, stream>>>(fin, bat, psum, pcnt);
  gmlp_kernel<<<1, 128, 0, stream>>>(psum, pcnt, gw1h, GM[1], gw2h, GM[3], out);
}
